// EnhancedTransformerBlock_78993038508550
// MI455X (gfx1250) — matrix operand order repaired and checked by reading, not compiled, not run on hardware
//
#include <hip/hip_runtime.h>
#include <math.h>

// ---------------------------------------------------------------------------
// CDNA5 / gfx1250 types
// ---------------------------------------------------------------------------
typedef __attribute__((ext_vector_type(16))) _Float16 v16h;
typedef __attribute__((ext_vector_type(8)))  _Float16 v8h;
typedef __attribute__((ext_vector_type(8)))  float    v8f;
typedef __attribute__((ext_vector_type(4)))  int      v4i;

__device__ __forceinline__ v8f wmma16(v16h a, v16h b, v8f c) {
  // D = A(16x32 f16) * B(32x16 f16) + C(16x16 f32)
  return __builtin_amdgcn_wmma_f32_16x16x32_f16(
      /*neg_a=*/false, a, /*neg_b=*/false, b,
      /*c_mod=*/(short)0, c, /*reuse_a=*/false, /*reuse_b=*/false);
}

// ---------------------------------------------------------------------------
// Optional CDNA5 async global->LDS staging (ASYNCcnt path), with fallback.
// Probe result (round 2 diagnostic): param 1 is `int4 addrspace(1)*` (global
// src), so signature is (global v4i*, lds v4i*, imm offset, imm cpol).
// ---------------------------------------------------------------------------
#if defined(__has_builtin)
#  if __has_builtin(__builtin_amdgcn_global_load_async_to_lds_b128)
#    define USE_ASYNC_LDS 1
#  endif
#endif
#ifndef USE_ASYNC_LDS
#  define USE_ASYNC_LDS 0
#endif

#if USE_ASYNC_LDS
typedef __attribute__((address_space(1))) v4i* gp_v4i;  // global int4*
typedef __attribute__((address_space(3))) v4i* lp_v4i;  // LDS int4*
__device__ __forceinline__ void async_cp16(void* lds, const void* gmem) {
  // 16-byte global -> LDS DMA, tracked with ASYNCcnt.
  __builtin_amdgcn_global_load_async_to_lds_b128(
      (gp_v4i)(unsigned long long)(uintptr_t)gmem,
      (lp_v4i)(unsigned int)(uintptr_t)lds,
      0, 0);
}
__device__ __forceinline__ void wait_async0() {
#  if __has_builtin(__builtin_amdgcn_s_wait_asynccnt)
  __builtin_amdgcn_s_wait_asynccnt(0);
#  else
  __asm__ volatile("s_wait_asynccnt 0" ::: "memory");
#  endif
}
#else
__device__ __forceinline__ void wait_async0() {}
#endif

// A fragment (16x32, row-major source, leading dim ld in halves).
// ISA layout: lane = {m = lane&15, half = lane>>4}; half 0: K=0..7 (v0..3),
// 16..23 (v4..7); half 1: +8.
__device__ __forceinline__ v16h frag_a(const _Float16* base, int ld, int kBase, int lane) {
  const int m = lane & 15, hf = lane >> 4;
  const _Float16* p = base + m * ld + kBase + 8 * hf;
  v8h lo = *(const v8h*)(p);        // K = kBase + 8*hf + 0..7
  v8h hi = *(const v8h*)(p + 16);   // K = kBase + 16 + 8*hf + 0..7
  v16h r;
#pragma unroll
  for (int i = 0; i < 8; ++i) { r[i] = lo[i]; r[i + 8] = hi[i]; }
  return r;
}

// B fragment (32x16) read from a TRANSPOSED [n][k] tile (ld in halves).
// ISA layout: lane = {n = lane&15, half = lane>>4}; half 0: K=0..7, 16..23; half 1: +8.
__device__ __forceinline__ v16h frag_b(const _Float16* base, int ld, int kBase, int lane) {
  const int n = lane & 15, hf = lane >> 4;
  const _Float16* p = base + n * ld + kBase + 8 * hf;
  v8h lo = *(const v8h*)(p);       // K = kBase + 8*hf + 0..7
  v8h hi = *(const v8h*)(p + 16);  // K = kBase + 16 + 8*hf + 0..7
  v16h r;
#pragma unroll
  for (int i = 0; i < 8; ++i) { r[i] = lo[i]; r[i + 8] = hi[i]; }
  return r;
}

// ---------------------------------------------------------------------------
// f32 -> f16 conversion (activations, straight copy)
// ---------------------------------------------------------------------------
__global__ void cvt_f32_f16(const float* __restrict__ in, _Float16* __restrict__ out, size_t n) {
  size_t i = (size_t)blockIdx.x * blockDim.x + threadIdx.x;
  size_t stride = (size_t)gridDim.x * blockDim.x;
  for (; i < n; i += stride) out[i] = (_Float16)in[i];
}

// f32 [K][N] -> f16 [N][K] (weights, one-time LDS-tiled transpose so the GEMM
// hot loop stages B with straight b128 copies / async DMA, no VALU transpose)
__global__ __launch_bounds__(256) void cvt_transpose_f16(
    const float* __restrict__ in, _Float16* __restrict__ out, int K, int N) {
  __shared__ float t[32][33];
  const int kb = blockIdx.y * 32, nb = blockIdx.x * 32;
  const int tx = threadIdx.x & 31, ty = threadIdx.x >> 5;  // 32 x 8
  for (int i = ty; i < 32; i += 8) t[i][tx] = in[(size_t)(kb + i) * N + nb + tx];
  __syncthreads();
  for (int i = ty; i < 32; i += 8)
    out[(size_t)(nb + i) * K + kb + tx] = (_Float16)t[tx][i];
}

// ---------------------------------------------------------------------------
// WMMA GEMM: C[MxN] = epi(A[MxK](f16) * Bt[NxK](f16)^T + bias) (+res)
// 256 threads = 8 waves; tile 128x128; wave = 64x32; BK = 32; double-buffered
// LDS staging (async DMA when available).
// ---------------------------------------------------------------------------
enum { EPI_NONE = 0, EPI_GELU = 1, EPI_RELU = 2, EPI_SIG = 3 };

template <int EPI, bool RES, bool WF32, bool WF16>
__global__ __launch_bounds__(256) void gemm_kernel(
    const _Float16* __restrict__ A, const _Float16* __restrict__ Bt,
    const float* __restrict__ bias, const float* __restrict__ res,
    float* __restrict__ Cf, _Float16* __restrict__ Ch,
    int M, int N, int K) {
  __shared__ __align__(16) _Float16 As[2][128 * 40];  // [m][k], padded ld=40
  __shared__ __align__(16) _Float16 Bs[2][128 * 40];  // [n][k], padded ld=40
  const int tid  = threadIdx.x;
  const int lane = tid & 31;
  const int wv   = tid >> 5;
  const int wm   = (wv >> 2) * 64;  // wave row offset (0/64)
  const int wn   = (wv & 3) * 32;   // wave col offset (0..96)
  const int mBase = blockIdx.y * 128;
  const int nBase = blockIdx.x * 128;

  // Both tiles are 128 rows x 32 halves: 512 b128 chunks, 2 per thread each.
  auto issue_stage = [&](int k0, int buf) {
    for (int ch = tid; ch < 512; ch += 256) {
      const int row = ch >> 2, kc = (ch & 3) * 8;
      const _Float16* ga = A  + (size_t)(mBase + row) * K + k0 + kc;
      const _Float16* gb = Bt + (size_t)(nBase + row) * K + k0 + kc;
      _Float16* la = &As[buf][row * 40 + kc];
      _Float16* lb = &Bs[buf][row * 40 + kc];
#if USE_ASYNC_LDS
      async_cp16(la, ga);
      async_cp16(lb, gb);
#else
      *(uint4*)la = *(const uint4*)ga;
      *(uint4*)lb = *(const uint4*)gb;
#endif
    }
  };

  v8f acc[4][2] = {};
  const int nst = K >> 5;

  issue_stage(0, 0);
  wait_async0();
  __syncthreads();

  for (int kt = 0; kt < nst; ++kt) {
    const int buf = kt & 1;
    if (kt + 1 < nst) issue_stage((kt + 1) << 5, buf ^ 1);  // overlap with WMMA

    v16h af[4], bf[2];
#pragma unroll
    for (int i = 0; i < 4; ++i) af[i] = frag_a(&As[buf][(wm + i * 16) * 40], 40, 0, lane);
#pragma unroll
    for (int j = 0; j < 2; ++j) bf[j] = frag_b(&Bs[buf][(wn + j * 16) * 40], 40, 0, lane);
#pragma unroll
    for (int i = 0; i < 4; ++i)
#pragma unroll
      for (int j = 0; j < 2; ++j) acc[i][j] = wmma16(af[i], bf[j], acc[i][j]);

    wait_async0();      // drain next-stage DMA before the barrier
    __syncthreads();
  }

  // --- epilogue: C layout m = vgpr + 8*half, n = lane&15
  const int hf = lane >> 4, nl = lane & 15;
#pragma unroll
  for (int j = 0; j < 2; ++j) {
    const int gn = nBase + wn + j * 16 + nl;
    const float bv = bias ? bias[gn] : 0.0f;
#pragma unroll
    for (int i = 0; i < 4; ++i) {
#pragma unroll
      for (int r = 0; r < 8; ++r) {
        const int gm = mBase + wm + i * 16 + r + 8 * hf;
        float v = acc[i][j][r] + bv;
        if (EPI == EPI_GELU)      v = 0.5f * v * (1.0f + erff(v * 0.70710678f));
        else if (EPI == EPI_RELU) v = fmaxf(v, 0.0f);
        else if (EPI == EPI_SIG)  v = 1.0f / (1.0f + __expf(-v));
        if (RES)  v += res[(size_t)gm * N + gn];
        if (WF32) Cf[(size_t)gm * N + gn] = v;
        if (WF16) Ch[(size_t)gm * N + gn] = (_Float16)v;
      }
    }
  }
}

// ---------------------------------------------------------------------------
// Flash attention: one (batch, head, 64-row q tile) per block; 4 waves,
// each wave owns 16 q rows. Online softmax; scores and P*V via WMMA.
// hd = 64, scale = 1/8. Fragments hoisted + batched so 8 WMMAs issue
// back-to-back behind a single ds wait.
// ---------------------------------------------------------------------------
__global__ __launch_bounds__(128) void attn_kernel(
    const _Float16* __restrict__ Qh, const _Float16* __restrict__ Kh,
    const _Float16* __restrict__ Vh, _Float16* __restrict__ Ctx,
    int S, int H) {
  __shared__ __align__(16) _Float16 Qs[64 * 72];      // [q][d]
  __shared__ __align__(16) _Float16 Ks[64 * 72];      // [key][d]   (B for scores)
  __shared__ __align__(16) _Float16 Vs[64 * 72];      // [d][key]   (B for P*V, transposed)
  __shared__ __align__(16) _Float16 Ps[4][16 * 72];   // per-wave P tile [q][key]
  const int tid = threadIdx.x, lane = tid & 31, wv = tid >> 5;
  const int qt = blockIdx.x, head = blockIdx.y, b = blockIdx.z;
  const size_t base = ((size_t)b * S) * H + (size_t)head * 64;

  for (int ch = tid; ch < 512; ch += 128) {
    const int row = ch >> 3, dc = (ch & 7) * 8;
    *(uint4*)(&Qs[row * 72 + dc]) =
        *(const uint4*)(Qh + base + (size_t)(qt * 64 + row) * H + dc);
  }
  float m8[8], l8[8];
#pragma unroll
  for (int r = 0; r < 8; ++r) { m8[r] = -3.0e38f; l8[r] = 0.0f; }
  v8f co[4] = {};
  __syncthreads();

  for (int kb = 0; kb < S; kb += 64) {
    for (int ch = tid; ch < 512; ch += 128) {
      const int row = ch >> 3, dc = (ch & 7) * 8;
      *(uint4*)(&Ks[row * 72 + dc]) =
          *(const uint4*)(Kh + base + (size_t)(kb + row) * H + dc);
      uint4 vv = *(const uint4*)(Vh + base + (size_t)(kb + row) * H + dc);
      __align__(16) _Float16 tmp[8];
      *(uint4*)tmp = vv;
#pragma unroll
      for (int j = 0; j < 8; ++j) Vs[(dc + j) * 72 + row] = tmp[j];  // transpose
    }
    // prefetch next K/V block (one 128B row per thread) -> global_prefetch_b8
    if (kb + 64 < S) {
      const int pr = tid & 63;
      __builtin_prefetch(Kh + base + (size_t)(kb + 64 + pr) * H, 0, 3);
      __builtin_prefetch(Vh + base + (size_t)(kb + 64 + pr) * H, 0, 3);
    }
    __syncthreads();

    // scores S = Q * K^T : 16 x 64, K-dim = hd = 64 (2 WMMA k-steps)
    v16h qa0 = frag_a(&Qs[(wv * 16) * 72], 72, 0, lane);
    v16h qa1 = frag_a(&Qs[(wv * 16) * 72], 72, 32, lane);
    v16h kbf[8];
#pragma unroll
    for (int nf = 0; nf < 4; ++nf) {
      kbf[nf]     = frag_b(&Ks[(nf * 16) * 72], 72, 0, lane);
      kbf[4 + nf] = frag_b(&Ks[(nf * 16) * 72], 72, 32, lane);
    }
    v8f sc[4] = {};
#pragma unroll
    for (int nf = 0; nf < 4; ++nf) sc[nf] = wmma16(qa0, kbf[nf], sc[nf]);
#pragma unroll
    for (int nf = 0; nf < 4; ++nf) sc[nf] = wmma16(qa1, kbf[4 + nf], sc[nf]);

    const float isc = 0.125f;  // 1/sqrt(64)
#pragma unroll
    for (int nf = 0; nf < 4; ++nf)
#pragma unroll
      for (int r = 0; r < 8; ++r) sc[nf][r] *= isc;

    // online softmax per row (row r+8*half lives in a 16-lane half-group)
#pragma unroll
    for (int r = 0; r < 8; ++r) {
      float mx = fmaxf(fmaxf(sc[0][r], sc[1][r]), fmaxf(sc[2][r], sc[3][r]));
#pragma unroll
      for (int msk = 8; msk; msk >>= 1) mx = fmaxf(mx, __shfl_xor(mx, msk, 32));
      const float mn = fmaxf(m8[r], mx);
      const float alpha = __expf(m8[r] - mn);
      float rs = 0.0f;
#pragma unroll
      for (int nf = 0; nf < 4; ++nf) {
        const float p = __expf(sc[nf][r] - mn);
        sc[nf][r] = p;
        rs += p;
      }
#pragma unroll
      for (int msk = 8; msk; msk >>= 1) rs += __shfl_xor(rs, msk, 32);
      l8[r] = l8[r] * alpha + rs;
      m8[r] = mn;
#pragma unroll
      for (int nf = 0; nf < 4; ++nf) co[nf][r] *= alpha;
    }

    // C-layout -> A-fragment layout via wave-private LDS round trip
    {
      const int hf = lane >> 4, nl = lane & 15;
#pragma unroll
      for (int nf = 0; nf < 4; ++nf)
#pragma unroll
        for (int r = 0; r < 8; ++r)
          Ps[wv][(r + 8 * hf) * 72 + nf * 16 + nl] = (_Float16)sc[nf][r];
    }
    __asm__ volatile("s_wait_dscnt 0" ::: "memory");  // wave-local LDS RAW

    // ctx += P(16x64) * V(64x64)
    v16h pa0 = frag_a(&Ps[wv][0], 72, 0, lane);
    v16h pa1 = frag_a(&Ps[wv][0], 72, 32, lane);
    v16h vbf[8];
#pragma unroll
    for (int nf = 0; nf < 4; ++nf) {
      vbf[nf]     = frag_b(&Vs[(nf * 16) * 72], 72, 0, lane);
      vbf[4 + nf] = frag_b(&Vs[(nf * 16) * 72], 72, 32, lane);
    }
#pragma unroll
    for (int nf = 0; nf < 4; ++nf) co[nf] = wmma16(pa0, vbf[nf], co[nf]);
#pragma unroll
    for (int nf = 0; nf < 4; ++nf) co[nf] = wmma16(pa1, vbf[4 + nf], co[nf]);

    __syncthreads();
  }

  const int hf = lane >> 4, nl = lane & 15;
#pragma unroll
  for (int r = 0; r < 8; ++r) {
    const float inv = 1.0f / l8[r];
    const int row = qt * 64 + wv * 16 + r + 8 * hf;
#pragma unroll
    for (int nf = 0; nf < 4; ++nf)
      Ctx[base + (size_t)row * H + nf * 16 + nl] = (_Float16)(co[nf][r] * inv);
  }
}

// ---------------------------------------------------------------------------
// Ada-LayerNorm: one row (H=1024) per 256-thread block.
// out = ((x-mu)*rstd*w + b) * (1 + sigmoid(vol*vsw+vsb)) * gamma + beta
// ---------------------------------------------------------------------------
__device__ __forceinline__ float block_sum(float v, float* sm) {
#pragma unroll
  for (int m = 16; m; m >>= 1) v += __shfl_xor(v, m, 32);
  const int wv = threadIdx.x >> 5, ln = threadIdx.x & 31;
  if (ln == 0) sm[wv] = v;
  __syncthreads();
  float t = (threadIdx.x < 8) ? sm[threadIdx.x] : 0.0f;
#pragma unroll
  for (int m = 4; m; m >>= 1) t += __shfl_xor(t, m, 32);
  if (threadIdx.x == 0) sm[0] = t;
  __syncthreads();
  const float r = sm[0];
  __syncthreads();
  return r;
}

__global__ __launch_bounds__(256) void ada_ln_kernel(
    const float* __restrict__ y, const float* __restrict__ vol,
    const float* __restrict__ lw, const float* __restrict__ lb,
    const float* __restrict__ gam, const float* __restrict__ bet,
    const float* __restrict__ vsw, const float* __restrict__ vsb,
    float* __restrict__ outf, _Float16* __restrict__ outh, int H) {
  __shared__ float sm[8];
  const int row = blockIdx.x;
  const float* xr = y + (size_t)row * H;
  float s = 0.0f;
  for (int i = threadIdx.x; i < H; i += 256) s += xr[i];
  const float mean = block_sum(s, sm) / (float)H;
  float v2 = 0.0f;
  for (int i = threadIdx.x; i < H; i += 256) { const float d = xr[i] - mean; v2 += d * d; }
  const float var = block_sum(v2, sm) / (float)H;
  const float rstd = rsqrtf(var + 1e-5f);
  const float z = vol[row] * vsw[0] + vsb[0];
  const float sg = 1.0f / (1.0f + __expf(-z));
  const float gmul = (1.0f + sg) * gam[0];
  const float badd = bet[0];
  for (int i = threadIdx.x; i < H; i += 256) {
    float v = (xr[i] - mean) * rstd * lw[i] + lb[i];
    v = v * gmul + badd;
    outf[(size_t)row * H + i] = v;
    if (outh) outh[(size_t)row * H + i] = (_Float16)v;
  }
}

// ---------------------------------------------------------------------------
// Gated FFN mix: f <- x1 + f*g + x1*(1-g)   (in-place on f)
// ---------------------------------------------------------------------------
__global__ void mix_kernel(const float* __restrict__ x1, float* __restrict__ f,
                           const float* __restrict__ g, size_t n) {
  size_t i = (size_t)blockIdx.x * blockDim.x + threadIdx.x;
  size_t stride = (size_t)gridDim.x * blockDim.x;
  for (; i < n; i += stride) {
    const float gg = g[i], xv = x1[i];
    f[i] = xv + f[i] * gg + xv * (1.0f - gg);
  }
}

// ---------------------------------------------------------------------------
// Host-side orchestration
// ---------------------------------------------------------------------------
extern "C" void kernel_launch(void* const* d_in, const int* in_sizes, int n_in,
                              void* d_out, int out_size, void* d_ws, size_t ws_size,
                              hipStream_t stream) {
  (void)n_in; (void)out_size; (void)ws_size; (void)in_sizes;
  const int Bsz = 8, S = 1024, H = 1024, Hf = 4096, Hg = 512;
  const int M = Bsz * S;  // 8192 tokens

  const float* x      = (const float*)d_in[0];
  const float* vol    = (const float*)d_in[1];
  const float* Wq     = (const float*)d_in[2];
  const float* bq     = (const float*)d_in[3];
  const float* Wk     = (const float*)d_in[4];
  const float* bk     = (const float*)d_in[5];
  const float* Wv     = (const float*)d_in[6];
  const float* bv     = (const float*)d_in[7];
  const float* Wo     = (const float*)d_in[8];
  const float* bo     = (const float*)d_in[9];
  const float* ln1_w  = (const float*)d_in[10];
  const float* ln1_b  = (const float*)d_in[11];
  const float* gamma1 = (const float*)d_in[12];
  const float* beta1  = (const float*)d_in[13];
  const float* vs1_w  = (const float*)d_in[14];
  const float* vs1_b  = (const float*)d_in[15];
  const float* ffn_w1 = (const float*)d_in[16];
  const float* ffn_b1 = (const float*)d_in[17];
  const float* ffn_w2 = (const float*)d_in[18];
  const float* ffn_b2 = (const float*)d_in[19];
  const float* gate_w1 = (const float*)d_in[20];
  const float* gate_b1 = (const float*)d_in[21];
  const float* gate_w2 = (const float*)d_in[22];
  const float* gate_b2 = (const float*)d_in[23];
  const float* ln2_w  = (const float*)d_in[24];
  const float* ln2_b  = (const float*)d_in[25];
  const float* gamma2 = (const float*)d_in[26];
  const float* beta2  = (const float*)d_in[27];
  const float* vs2_w  = (const float*)d_in[28];
  const float* vs2_b  = (const float*)d_in[29];

  // Workspace layout (bump-allocated, with reuse). MB = 2^20 bytes.
  // Weight buffers hold TRANSPOSED f16 weights [N][K].
  char* w = (char*)d_ws;
  const size_t MB = 1ull << 20;
  _Float16* xh  = (_Float16*)(w + 0);        // 16 MB  [M,H]
  _Float16* Wqh = (_Float16*)(w + 16 * MB);  // 2 MB each, [H][H]^T
  _Float16* Wkh = (_Float16*)(w + 18 * MB);
  _Float16* Wvh = (_Float16*)(w + 20 * MB);
  _Float16* Woh = (_Float16*)(w + 22 * MB);
  _Float16* W1h = (_Float16*)(w + 24 * MB);  // 8 MB [Hf][H]
  _Float16* W2h = (_Float16*)(w + 32 * MB);  // 8 MB [H][Hf]
  _Float16* G1h = (_Float16*)(w + 40 * MB);  // 1 MB [Hg][H]
  _Float16* G2h = (_Float16*)(w + 41 * MB);  // 1 MB [H][Hg]
  _Float16* Qh  = (_Float16*)(w + 42 * MB);  // 16 MB
  _Float16* Kh  = (_Float16*)(w + 58 * MB);  // 16 MB
  _Float16* Vh  = (_Float16*)(w + 74 * MB);  // 16 MB
  _Float16* Ch  = (_Float16*)(w + 90 * MB);  // 16 MB
  _Float16* Hh  = (_Float16*)(w + 42 * MB);  // 64 MB (reuses Q/K/V/Ctx after Wo GEMM)
  float*    y1  = (float*)(w + 106 * MB);    // 32 MB  x + attn_out
  float*    ffn = (float*)(w + 106 * MB);    // 32 MB  (reuses y1 after adaLN1)
  float*    x1  = (float*)(w + 138 * MB);    // 32 MB
  _Float16* x1h = (_Float16*)(w + 170 * MB); // 16 MB
  _Float16* gh  = (_Float16*)(w + 186 * MB); // 8 MB  [M,Hg]
  float*    gf  = (float*)(w + 194 * MB);    // 32 MB -> total 226 MB

  const size_t nMH = (size_t)M * H;

  // --- 0. conversions: activations straight, weights transposed to [N][K]
  cvt_f32_f16<<<2048, 256, 0, stream>>>(x, xh, nMH);
  cvt_transpose_f16<<<dim3(H / 32, H / 32), 256, 0, stream>>>(Wq, Wqh, H, H);
  cvt_transpose_f16<<<dim3(H / 32, H / 32), 256, 0, stream>>>(Wk, Wkh, H, H);
  cvt_transpose_f16<<<dim3(H / 32, H / 32), 256, 0, stream>>>(Wv, Wvh, H, H);
  cvt_transpose_f16<<<dim3(H / 32, H / 32), 256, 0, stream>>>(Wo, Woh, H, H);
  cvt_transpose_f16<<<dim3(Hf / 32, H / 32), 256, 0, stream>>>(ffn_w1, W1h, H, Hf);
  cvt_transpose_f16<<<dim3(H / 32, Hf / 32), 256, 0, stream>>>(ffn_w2, W2h, Hf, H);
  cvt_transpose_f16<<<dim3(Hg / 32, H / 32), 256, 0, stream>>>(gate_w1, G1h, H, Hg);
  cvt_transpose_f16<<<dim3(H / 32, Hg / 32), 256, 0, stream>>>(gate_w2, G2h, Hg, H);

  // --- 1. Q/K/V projections (f16 out, f32 accum)
  dim3 gHH(H / 128, M / 128);
  gemm_kernel<EPI_NONE, false, false, true><<<gHH, 256, 0, stream>>>(xh, Wqh, bq, nullptr, nullptr, Qh, M, H, H);
  gemm_kernel<EPI_NONE, false, false, true><<<gHH, 256, 0, stream>>>(xh, Wkh, bk, nullptr, nullptr, Kh, M, H, H);
  gemm_kernel<EPI_NONE, false, false, true><<<gHH, 256, 0, stream>>>(xh, Wvh, bv, nullptr, nullptr, Vh, M, H, H);

  // --- 2. flash attention (16 q-tiles x 16 heads x 8 batches)
  attn_kernel<<<dim3(S / 64, 16, Bsz), 128, 0, stream>>>(Qh, Kh, Vh, Ch, S, H);

  // --- 3. output projection, fused residual: y1 = x + ctx@Wo + bo
  gemm_kernel<EPI_NONE, true, true, false><<<gHH, 256, 0, stream>>>(Ch, Woh, bo, x, y1, nullptr, M, H, H);

  // --- 4. adaLN1 -> x1 (f32) + x1h (f16)
  ada_ln_kernel<<<M, 256, 0, stream>>>(y1, vol, ln1_w, ln1_b, gamma1, beta1, vs1_w, vs1_b, x1, x1h, H);

  // --- 5. FFN up + exact GELU (f16 out)
  gemm_kernel<EPI_GELU, false, false, true><<<dim3(Hf / 128, M / 128), 256, 0, stream>>>(
      x1h, W1h, ffn_b1, nullptr, nullptr, Hh, M, Hf, H);
  // --- 6. FFN down (f32 out)
  gemm_kernel<EPI_NONE, false, true, false><<<gHH, 256, 0, stream>>>(
      Hh, W2h, ffn_b2, nullptr, ffn, nullptr, M, H, Hf);

  // --- 7. gate: relu(x1@G1+b) then sigmoid(...@G2+b)
  gemm_kernel<EPI_RELU, false, false, true><<<dim3(Hg / 128, M / 128), 256, 0, stream>>>(
      x1h, G1h, gate_b1, nullptr, nullptr, gh, M, Hg, H);
  gemm_kernel<EPI_SIG, false, true, false><<<gHH, 256, 0, stream>>>(
      gh, G2h, gate_b2, nullptr, gf, nullptr, M, H, Hg);

  // --- 8. gated mix + residual (in-place on ffn buffer)
  mix_kernel<<<2048, 256, 0, stream>>>(x1, ffn, gf, nMH);

  // --- 9. adaLN2 -> final output (f32)
  ada_ln_kernel<<<M, 256, 0, stream>>>(ffn, vol, ln2_w, ln2_b, gamma2, beta2, vs2_w, vs2_b,
                                       (float*)d_out, nullptr, H);
}
